// GatLayer_76081050682031
// MI455X (gfx1250) — matrix operand order repaired and checked by reading, not compiled, not run on hardware
//
#include <hip/hip_runtime.h>
#include <hip/hip_bf16.h>

#define N_NODES 10000
#define N_EDGES 160000
#define IN_FEATS 512
#define HEADS 8
#define OUT_FEATS 64
#define HF 512              // HEADS*OUT_FEATS
#define NEG_SLOPE 0.2f
#define M_TILES 625         // 10000/16 exact
#define N_TILES 32          // 512/16

typedef __attribute__((ext_vector_type(16))) __bf16 v16bf;
typedef __attribute__((ext_vector_type(8)))  float  v8f;

// ---------- helpers ----------
__device__ __forceinline__ __bf16 f2bf(float f) {
    unsigned u = __float_as_uint(f);
    unsigned r = (u + 0x7FFFu + ((u >> 16) & 1u)) >> 16;   // RNE
    unsigned short h = (unsigned short)r;
    __bf16 b;
    __builtin_memcpy(&b, &h, 2);
    return b;
}
__device__ __forceinline__ float bf2f(__bf16 b) {
    unsigned short h;
    __builtin_memcpy(&h, &b, 2);
    return __uint_as_float(((unsigned)h) << 16);
}
// order-preserving float<->uint for atomic max
__device__ __forceinline__ unsigned fflip(float f) {
    unsigned u = __float_as_uint(f);
    return (u & 0x80000000u) ? ~u : (u | 0x80000000u);
}
__device__ __forceinline__ float funflip(unsigned u) {
    return __uint_as_float((u & 0x80000000u) ? (u & 0x7FFFFFFFu) : ~u);
}

// ---------- kernel 0: init out<-bias, max keys<-0, sums<-0 ----------
__global__ void k_init(float* __restrict__ out, const float* __restrict__ bias,
                       unsigned* __restrict__ mkey, float* __restrict__ ssum) {
    int id = blockIdx.x * blockDim.x + threadIdx.x;
    if (id < N_NODES * HF) out[id] = bias[id & (HF - 1)];
    if (id < N_NODES * HEADS) { mkey[id] = 0u; ssum[id] = 0.0f; }
}

// ---------- kernel 1: split feat into bf16 hi/lo ----------
__global__ void k_cvt_feat(const float* __restrict__ x,
                           __bf16* __restrict__ hi, __bf16* __restrict__ lo) {
    int id = blockIdx.x * blockDim.x + threadIdx.x;
    if (id >= N_NODES * IN_FEATS) return;
    float f = x[id];
    __bf16 h = f2bf(f);
    hi[id] = h;
    lo[id] = f2bf(f - bf2f(h));
}

// ---------- kernel 2: split + transpose W -> Wt[n*512+k] bf16 hi/lo ----------
__global__ void k_cvt_wt(const float* __restrict__ W,
                         __bf16* __restrict__ hi, __bf16* __restrict__ lo) {
    int id = blockIdx.x * blockDim.x + threadIdx.x;   // id = n*512 + k
    if (id >= IN_FEATS * HF) return;
    int n = id >> 9, k = id & 511;
    float f = W[k * HF + n];
    __bf16 h = f2bf(f);
    hi[id] = h;
    lo[id] = f2bf(f - bf2f(h));
}

// ---------- kernel 3: WMMA GEMM  h = feat @ W  (split-bf16, 3 wmma/K-step) ----------
__device__ __forceinline__ v16bf ldA(const __bf16* p) {
    union { v16bf v; uint4 q[2]; } u;
    u.q[0] = *reinterpret_cast<const uint4*>(p);        // K block +0..7
    u.q[1] = *reinterpret_cast<const uint4*>(p + 8);    // K block +8..15
    return u.v;
}

__global__ __launch_bounds__(256)
void k_gemm(const __bf16* __restrict__ Ah, const __bf16* __restrict__ Al,
            const __bf16* __restrict__ Bh, const __bf16* __restrict__ Bl,
            float* __restrict__ H) {
    const int wave = blockIdx.x * (blockDim.x >> 5) + (threadIdx.x >> 5);
    if (wave >= M_TILES * N_TILES) return;              // uniform per wave
    const int lane  = threadIdx.x & 31;
    const int tileM = wave >> 5;                        // /N_TILES
    const int tileN = wave & 31;
    const int half  = lane >> 4;
    const int l15   = lane & 15;
    const size_t arow = (size_t)((tileM << 4) + l15) * IN_FEATS;
    const int    ncol = (tileN << 4) + l15;
    const size_t bcol = (size_t)ncol * IN_FEATS;

    v8f c = {};
#pragma unroll 2
    for (int k0 = 0; k0 < IN_FEATS; k0 += 32) {
        const int aoff = k0 + half * 16;
        const int boff = k0 + half * 16;
        v16bf ah = ldA(Ah + arow + aoff);
        v16bf al = ldA(Al + arow + aoff);
        v16bf bh = *reinterpret_cast<const v16bf*>(Bh + bcol + boff);
        v16bf bl = *reinterpret_cast<const v16bf*>(Bl + bcol + boff);
        c = __builtin_amdgcn_wmma_f32_16x16x32_bf16(false, al, false, bh, (short)0, c, false, false);
        c = __builtin_amdgcn_wmma_f32_16x16x32_bf16(false, ah, false, bl, (short)0, c, false, false);
        c = __builtin_amdgcn_wmma_f32_16x16x32_bf16(false, ah, false, bh, (short)0, c, false, false);
    }
    const int row0 = (tileM << 4) + half * 8;
#pragma unroll
    for (int r = 0; r < 8; ++r)
        H[(size_t)(row0 + r) * HF + ncol] = c[r];
}

// ---------- kernel 4: per-node logits el/er ----------
__global__ void k_logits(const float* __restrict__ H,
                         const float* __restrict__ al, const float* __restrict__ ar,
                         float* __restrict__ el, float* __restrict__ er) {
    int id = blockIdx.x * blockDim.x + threadIdx.x;
    if (id >= N_NODES * HEADS) return;
    int n = id >> 3, hh = id & 7;
    const float* hp = H + (size_t)n * HF + hh * OUT_FEATS;
    const float* lp = al + hh * OUT_FEATS;
    const float* rp = ar + hh * OUT_FEATS;
    float sl = 0.f, sr = 0.f;
#pragma unroll 4
    for (int f = 0; f < OUT_FEATS; ++f) { float v = hp[f]; sl += v * lp[f]; sr += v * rp[f]; }
    el[id] = sl; er[id] = sr;
}

// ---------- kernel 5: edge score + LeakyReLU + segment max ----------
__global__ void k_edge_score(const int* __restrict__ src, const int* __restrict__ dst,
                             const float* __restrict__ el, const float* __restrict__ er,
                             float* __restrict__ esc, unsigned* __restrict__ mkey) {
    int id = blockIdx.x * blockDim.x + threadIdx.x;
    if (id >= N_EDGES * HEADS) return;
    int e = id >> 3, hh = id & 7;
    float v = el[src[e] * HEADS + hh] + er[dst[e] * HEADS + hh];
    v = v > 0.f ? v : NEG_SLOPE * v;
    esc[id] = v;
    atomicMax(&mkey[dst[e] * HEADS + hh], fflip(v));
}

// ---------- kernel 6: exp(e - max) + segment sum ----------
__global__ void k_edge_exp(const int* __restrict__ dst,
                           float* __restrict__ esc, const unsigned* __restrict__ mkey,
                           float* __restrict__ ssum) {
    int id = blockIdx.x * blockDim.x + threadIdx.x;
    if (id >= N_EDGES * HEADS) return;
    int e = id >> 3, hh = id & 7;
    float m = funflip(mkey[dst[e] * HEADS + hh]);
    float a = __expf(esc[id] - m);
    esc[id] = a;
    atomicAdd(&ssum[dst[e] * HEADS + hh], a);
}

// ---------- kernel 7: weighted scatter aggregation (one block per edge) ----------
__global__ __launch_bounds__(256)
void k_aggregate(const int* __restrict__ src, const int* __restrict__ dst,
                 const float* __restrict__ esc, const float* __restrict__ ssum,
                 const float* __restrict__ H, float* __restrict__ out) {
    __shared__ float coef[HEADS];
    const int e = blockIdx.x;
    const int s = src[e], d = dst[e];
    if (threadIdx.x < HEADS)
        coef[threadIdx.x] = esc[e * HEADS + threadIdx.x] / ssum[d * HEADS + threadIdx.x];
    __syncthreads();
    const float* hp = H + (size_t)s * HF;
    float* op = out + (size_t)d * HF;
#pragma unroll
    for (int c = threadIdx.x; c < HF; c += 256)
        atomicAdd(&op[c], coef[c >> 6] * hp[c]);
}

// ---------- launch ----------
extern "C" void kernel_launch(void* const* d_in, const int* in_sizes, int n_in,
                              void* d_out, int out_size, void* d_ws, size_t ws_size,
                              hipStream_t stream) {
    const float* feat   = (const float*)d_in[0];
    const float* W      = (const float*)d_in[1];
    const float* attn_l = (const float*)d_in[2];
    const float* attn_r = (const float*)d_in[3];
    const float* bias   = (const float*)d_in[4];
    const int*   src    = (const int*)d_in[5];
    const int*   dst    = (const int*)d_in[6];
    float* out = (float*)d_out;

    char* ws = (char*)d_ws;
    size_t off = 0;
    float*    H      = (float*)(ws + off);  off += (size_t)N_NODES * HF * 4;        // 20.48 MB
    __bf16*   featHi = (__bf16*)(ws + off); off += (size_t)N_NODES * IN_FEATS * 2;  // 10.24 MB
    __bf16*   featLo = (__bf16*)(ws + off); off += (size_t)N_NODES * IN_FEATS * 2;
    __bf16*   WtHi   = (__bf16*)(ws + off); off += (size_t)IN_FEATS * HF * 2;       // 0.52 MB
    __bf16*   WtLo   = (__bf16*)(ws + off); off += (size_t)IN_FEATS * HF * 2;
    float*    el     = (float*)(ws + off);  off += (size_t)N_NODES * HEADS * 4;
    float*    er     = (float*)(ws + off);  off += (size_t)N_NODES * HEADS * 4;
    unsigned* mkey   = (unsigned*)(ws + off); off += (size_t)N_NODES * HEADS * 4;
    float*    ssum   = (float*)(ws + off);  off += (size_t)N_NODES * HEADS * 4;
    float*    esc    = (float*)(ws + off);  off += (size_t)N_EDGES * HEADS * 4;     // 5.12 MB

    const int B = 256;
    k_init<<<(N_NODES * HF + B - 1) / B, B, 0, stream>>>(out, bias, mkey, ssum);
    k_cvt_feat<<<(N_NODES * IN_FEATS + B - 1) / B, B, 0, stream>>>(feat, featHi, featLo);
    k_cvt_wt<<<(IN_FEATS * HF + B - 1) / B, B, 0, stream>>>(W, WtHi, WtLo);
    k_gemm<<<(M_TILES * N_TILES) / 8, B, 0, stream>>>(featHi, featLo, WtHi, WtLo, H);
    k_logits<<<(N_NODES * HEADS + B - 1) / B, B, 0, stream>>>(H, attn_l, attn_r, el, er);
    k_edge_score<<<(N_EDGES * HEADS + B - 1) / B, B, 0, stream>>>(src, dst, el, er, esc, mkey);
    k_edge_exp<<<(N_EDGES * HEADS + B - 1) / B, B, 0, stream>>>(dst, esc, mkey, ssum);
    k_aggregate<<<N_EDGES, B, 0, stream>>>(src, dst, esc, ssum, H, out);
}
